// LSTMEncoder_50465865728011
// MI455X (gfx1250) — compile-verified, matrix operand order checked by reading, not run on hardware
//
#include <hip/hip_runtime.h>
#include <hip/hip_bf16.h>

typedef __attribute__((ext_vector_type(16))) _Float16 v16h;
typedef __attribute__((ext_vector_type(8)))  _Float16 v8h;
typedef __attribute__((ext_vector_type(4)))  _Float16 v4h;
typedef __attribute__((ext_vector_type(2)))  _Float16 v2h;
typedef __attribute__((ext_vector_type(8)))  float    v8f;
typedef __attribute__((ext_vector_type(8)))  int      v8i;

namespace {
constexpr int T_SEQ = 4096;
constexpr int IN_DIM = 4;
constexpr int H1 = 64;          // layer-1 hidden
constexpr int H2 = 32;          // layer-2 hidden (latent)
constexpr int G1S = 260;        // padded f32 row stride for gates1 [16 x 256]
constexpr int G2S = 132;        // padded f32 row stride for gates2 [16 x 128]
constexpr int H1S = 72;         // padded f16 row stride for h1 (144 B)
constexpr int H2S = 40;         // padded f16 row stride for h2 (80 B)
constexpr int TCH = 256;        // timesteps staged per x-refill chunk
constexpr int XROWS = 17;       // padded rows per step in x tile (bank spread)
}

__device__ __forceinline__ v8f wmma_f16(v16h a, v16h b, v8f c) {
  // D = A(16x32 f16) * B(32x16 f16) + C(16x16 f32)
  return __builtin_amdgcn_wmma_f32_16x16x32_f16(false, a, false, b, (short)0, c,
                                                false, false);
}

// Pin a packed f16 tile into a contiguous VGPR tuple and make it opaque so the
// register allocator cannot re-materialize it from f32 sources inside the loop.
__device__ __forceinline__ v16h pin16h(v16h x) {
  union { v16h h; v8i i; } u;
  u.h = x;
  asm volatile("" : "+v"(u.i));
  return u.h;
}
__device__ __forceinline__ v8f pin8f(v8f x) {
  asm volatile("" : "+v"(x));
  return x;
}

// Load one 32x16 B-tile (f16) from a row-major f32 weight matrix W[N x ldk].
// B column n == W row n; CDNA5 16-bit operand K layout:
//   lanes 0-15 : K = kc+{0..7, 16..23};  lanes 16-31 : K = kc+{8..15, 24..31}
__device__ __forceinline__ v16h load_btile(const float* W, int ldk, int row,
                                           int kc, int kmax, int hi) {
  v16h b;
#pragma unroll
  for (int p = 0; p < 16; ++p) {
    int k = kc + hi * 8 + (p & 7) + ((p >> 3) << 4);
    float w = (k < kmax) ? W[row * ldk + k] : 0.0f;
    b[p] = (_Float16)w;
  }
  return pin16h(b);
}

// Build a 16x32 A-tile (f16) from an LDS f16 state buffer (row m per lane).
// The per-lane K slots are two contiguous 8-half (16B) runs -> ds_load_b128.
__device__ __forceinline__ v16h load_atile(const _Float16* hb, int stride,
                                           int m, int kc, int hi) {
  union { v16h v; v8h h[2]; } u;
  const _Float16* p0 = hb + m * stride + kc + hi * 8;
  u.h[0] = *(const v8h*)(p0);
  u.h[1] = *(const v8h*)(p0 + 16);
  return u.v;
}

__device__ __forceinline__ float fsig(float x) {
  return __builtin_amdgcn_rcpf(1.0f + __expf(-x));
}
__device__ __forceinline__ float ftanhf(float x) {
  return 1.0f - 2.0f * __builtin_amdgcn_rcpf(__expf(2.0f * x) + 1.0f);
}

__global__ __launch_bounds__(256, 1) void lstm2_scan_kernel(
    const float* __restrict__ x,      // [64, 4096, 4]
    const float* __restrict__ W_ih1,  // [256, 4]
    const float* __restrict__ W_hh1,  // [256, 64]
    const float* __restrict__ b_ih1,  // [256]
    const float* __restrict__ b_hh1,  // [256]
    const float* __restrict__ W_ih2,  // [128, 64]
    const float* __restrict__ W_hh2,  // [128, 32]
    const float* __restrict__ b_ih2,  // [128]
    const float* __restrict__ b_hh2,  // [128]
    float* __restrict__ out) {        // [64, 4096, 32]
  const int tid  = threadIdx.x;
  const int lane = tid & 31;
  const int wave = tid >> 5;        // 8 waves
  const int col  = lane & 15;       // N column within tile / A row M
  const int hi   = lane >> 4;       // half-lane selector for K layout
  const int b0   = blockIdx.x * 16; // this WG owns batch rows [b0, b0+16)

  __shared__ __align__(16) float    gates1[16 * G1S];
  __shared__ __align__(16) float    gates2[16 * G2S];
  // double-buffered recurrent state (parity): W-phase reads never collide
  // with P-phase writes -> only 2 barriers per timestep
  __shared__ __align__(16) _Float16 h1buf[2][16 * H1S];
  __shared__ __align__(16) _Float16 h2buf[2][16 * H2S];
  // staged x chunk, pre-converted to f16: [TCH steps][XROWS rows][4 halves]
  __shared__ __align__(16) _Float16 xlds[TCH * XROWS * 4];

  // ---- per-wave tile assignment --------------------------------------
  const int n1a = wave * 16;        // layer-1 gate cols [n1a, n1a+16)
  const int n1b = (wave + 8) * 16;  // layer-1 gate cols [n1b, n1b+16)
  const int n2  = wave * 16;        // layer-2 gate cols [n2,  n2 +16)

  // ---- preload all weights into registers (B-matrix layout, f16) -----
  v16h Bx_a  = load_btile(W_ih1, IN_DIM, n1a + col, 0,  IN_DIM, hi);
  v16h Bx_b  = load_btile(W_ih1, IN_DIM, n1b + col, 0,  IN_DIM, hi);
  v16h Bh_a0 = load_btile(W_hh1, H1,     n1a + col, 0,  H1,     hi);
  v16h Bh_a1 = load_btile(W_hh1, H1,     n1a + col, 32, H1,     hi);
  v16h Bh_b0 = load_btile(W_hh1, H1,     n1b + col, 0,  H1,     hi);
  v16h Bh_b1 = load_btile(W_hh1, H1,     n1b + col, 32, H1,     hi);
  v16h Bi2_0 = load_btile(W_ih2, H1,     n2  + col, 0,  H1,     hi);
  v16h Bi2_1 = load_btile(W_ih2, H1,     n2  + col, 32, H1,     hi);
  v16h Bh2_0 = load_btile(W_hh2, H2,     n2  + col, 0,  H2,     hi);

  // loop-invariant bias vectors used directly as WMMA C operands
  const float bias1a = b_ih1[n1a + col] + b_hh1[n1a + col];
  const float bias1b = b_ih1[n1b + col] + b_hh1[n1b + col];
  const float bias2  = b_ih2[n2  + col] + b_hh2[n2  + col];
  v8f biasv_a, biasv_b, biasv_2;
#pragma unroll
  for (int r = 0; r < 8; ++r) {
    biasv_a[r] = bias1a; biasv_b[r] = bias1b; biasv_2[r] = bias2;
  }
  biasv_a = pin8f(biasv_a);
  biasv_b = pin8f(biasv_b);
  biasv_2 = pin8f(biasv_2);

  // loop-invariant LDS gate-store bases (row = hi*8, advance by r*stride)
  float* g1a = &gates1[(hi * 8) * G1S + n1a + col];
  float* g1b = &gates1[(hi * 8) * G1S + n1b + col];
  float* g2p = &gates2[(hi * 8) * G2S + n2  + col];

  // ---- element-wise updater assignment (fixed per thread) ------------
  const int eb  = tid >> 4;         // batch row 0..15 (also x-refill row)
  const int ech = tid & 15;         // x-refill step sub-chunk
  const int ej1 = ech * 4;          // layer-1 hidden cols [ej1, ej1+4)
  const int ej2 = ech * 2;          // layer-2 hidden cols [ej2, ej2+2)
  float c1r[4] = {0.f, 0.f, 0.f, 0.f};
  float c2r[2] = {0.f, 0.f};
  float* outbase = out + ((size_t)(b0 + eb) * T_SEQ) * H2 + ej2;

  // ---- zero recurrent state buffers (both parities) -------------------
  for (int i = tid; i < 2 * 16 * H1S; i += 256) h1buf[0][i] = (_Float16)0.0f;
  for (int i = tid; i < 2 * 16 * H2S; i += 256) h2buf[0][i] = (_Float16)0.0f;

  const int m = col;  // A-matrix row for this lane

  // ===== fused scan: layer-1 at time t, layer-2 lagging at time t-1 =====
  for (int tb = 0; tb < T_SEQ; tb += TCH) {
    // ---- x refill: stage TCH steps of x into LDS as f16 ---------------
    // thread (eb=row, ech=sub-chunk) copies steps ts = ech + 16*i
    {
      const float* xsrc = x + ((size_t)(b0 + eb) * T_SEQ + tb + ech) * IN_DIM;
#pragma unroll 4
      for (int i = 0; i < TCH / 16; ++i) {
        float4 xv = *(const float4*)(xsrc + (size_t)(16 * i) * IN_DIM);
        v4h xh;
        xh[0] = (_Float16)xv.x; xh[1] = (_Float16)xv.y;
        xh[2] = (_Float16)xv.z; xh[3] = (_Float16)xv.w;
        *(v4h*)(&xlds[((ech + 16 * i) * XROWS + eb) * 4]) = xh;
      }
    }
    __syncthreads();

    for (int tt = 0; tt < TCH; ++tt) {
      const int t = tb + tt;
      // -------------- W-phase: all WMMAs ------------------------------
      // h1_{t-1} is in parity (t-1)&1 == (t+1)&1 ; h2_{t-2} in parity t&1
      const _Float16* h1prev = h1buf[(t + 1) & 1];
      const _Float16* h2prev = h2buf[t & 1];
      v16h ah0 = load_atile(h1prev, H1S, m, 0,  hi);  // shared by L1 + L2
      v16h ah1 = load_atile(h1prev, H1S, m, 32, hi);

      {  // layer-1 gates at time t (always: t < T_SEQ inside chunks)
        uint2 xd = *(const uint2*)(&xlds[(tt * XROWS + m) * 4]);
        union { v16h v; unsigned u[8]; } uax;
        uax.u[0] = hi ? 0u : xd.x;
        uax.u[1] = hi ? 0u : xd.y;
#pragma unroll
        for (int p = 2; p < 8; ++p) uax.u[p] = 0u;

        v8f acc_a = wmma_f16(uax.v, Bx_a,  biasv_a);
        acc_a     = wmma_f16(ah0,   Bh_a0, acc_a);
        acc_a     = wmma_f16(ah1,   Bh_a1, acc_a);
        v8f acc_b = wmma_f16(uax.v, Bx_b,  biasv_b);
        acc_b     = wmma_f16(ah0,   Bh_b0, acc_b);
        acc_b     = wmma_f16(ah1,   Bh_b1, acc_b);
#pragma unroll
        for (int r = 0; r < 8; ++r) {
          g1a[r * G1S] = acc_a[r];
          g1b[r * G1S] = acc_b[r];
        }
      }
      if (t >= 1) {  // layer-2 gates at time t-1
        v16h ah2 = load_atile(h2prev, H2S, m, 0, hi);   // h2_{t-2}
        v8f acc2 = wmma_f16(ah2, Bh2_0, biasv_2);
        acc2     = wmma_f16(ah0, Bi2_0, acc2);          // h1_{t-1} @ W_ih2^T
        acc2     = wmma_f16(ah1, Bi2_1, acc2);
#pragma unroll
        for (int r = 0; r < 8; ++r)
          g2p[r * G2S] = acc2[r];
      }
      __syncthreads();

      // -------------- P-phase: both element-wise updates --------------
      {  // layer-1 cell update at time t -> h1buf[t&1]
        const float* g = gates1 + eb * G1S;
        float4 gi = *(const float4*)(g + 0 * H1 + ej1);
        float4 gf = *(const float4*)(g + 1 * H1 + ej1);
        float4 gg = *(const float4*)(g + 2 * H1 + ej1);
        float4 go = *(const float4*)(g + 3 * H1 + ej1);
        float iv[4] = {gi.x, gi.y, gi.z, gi.w};
        float fv[4] = {gf.x, gf.y, gf.z, gf.w};
        float gv[4] = {gg.x, gg.y, gg.z, gg.w};
        float ov[4] = {go.x, go.y, go.z, go.w};
        v4h hq;
#pragma unroll
        for (int q = 0; q < 4; ++q) {
          float ig = fsig(iv[q]), fg = fsig(fv[q]);
          float gt = ftanhf(gv[q]), og = fsig(ov[q]);
          c1r[q] = fg * c1r[q] + ig * gt;
          hq[q] = (_Float16)(og * ftanhf(c1r[q]));
        }
        *(v4h*)(&h1buf[t & 1][eb * H1S + ej1]) = hq;
      }
      if (t >= 1) {  // layer-2 cell update at time t-1 -> h2buf[(t-1)&1]
        const float* g = gates2 + eb * G2S;
        float2 gi = *(const float2*)(g + 0 * H2 + ej2);
        float2 gf = *(const float2*)(g + 1 * H2 + ej2);
        float2 gg = *(const float2*)(g + 2 * H2 + ej2);
        float2 go = *(const float2*)(g + 3 * H2 + ej2);
        float iv[2] = {gi.x, gi.y}, fv[2] = {gf.x, gf.y};
        float gv[2] = {gg.x, gg.y}, ov[2] = {go.x, go.y};
        float2 ho;
        v2h h2q;
#pragma unroll
        for (int q = 0; q < 2; ++q) {
          float ig = fsig(iv[q]), fg = fsig(fv[q]);
          float gt = ftanhf(gv[q]), og = fsig(ov[q]);
          c2r[q] = fg * c2r[q] + ig * gt;
          float hv = og * ftanhf(c2r[q]);
          h2q[q] = (_Float16)hv;
          if (q == 0) ho.x = hv; else ho.y = hv;
        }
        *(v2h*)(&h2buf[(t - 1) & 1][eb * H2S + ej2]) = h2q;
        *(float2*)(outbase + (size_t)(t - 1) * H2) = ho;  // out[b, t-1, :]
      }
      __syncthreads();
    }
  }

  // ===== epilogue: lagged layer-2 step for t = T_SEQ (time T_SEQ-1) =====
  {
    const int t = T_SEQ;
    const _Float16* h1prev = h1buf[(t + 1) & 1];
    const _Float16* h2prev = h2buf[t & 1];
    v16h ah0 = load_atile(h1prev, H1S, m, 0,  hi);
    v16h ah1 = load_atile(h1prev, H1S, m, 32, hi);
    v16h ah2 = load_atile(h2prev, H2S, m, 0,  hi);
    v8f acc2 = wmma_f16(ah2, Bh2_0, biasv_2);
    acc2     = wmma_f16(ah0, Bi2_0, acc2);
    acc2     = wmma_f16(ah1, Bi2_1, acc2);
#pragma unroll
    for (int r = 0; r < 8; ++r)
      g2p[r * G2S] = acc2[r];
    __syncthreads();

    const float* g = gates2 + eb * G2S;
    float2 gi = *(const float2*)(g + 0 * H2 + ej2);
    float2 gf = *(const float2*)(g + 1 * H2 + ej2);
    float2 gg = *(const float2*)(g + 2 * H2 + ej2);
    float2 go = *(const float2*)(g + 3 * H2 + ej2);
    float iv[2] = {gi.x, gi.y}, fv[2] = {gf.x, gf.y};
    float gv[2] = {gg.x, gg.y}, ov[2] = {go.x, go.y};
    float2 ho;
#pragma unroll
    for (int q = 0; q < 2; ++q) {
      float ig = fsig(iv[q]), fg = fsig(fv[q]);
      float gt = ftanhf(gv[q]), og = fsig(ov[q]);
      c2r[q] = fg * c2r[q] + ig * gt;
      float hv = og * ftanhf(c2r[q]);
      if (q == 0) ho.x = hv; else ho.y = hv;
    }
    *(float2*)(outbase + (size_t)(t - 1) * H2) = ho;
  }
}

extern "C" void kernel_launch(void* const* d_in, const int* in_sizes, int n_in,
                              void* d_out, int out_size, void* d_ws, size_t ws_size,
                              hipStream_t stream) {
  const float* x     = (const float*)d_in[0];
  const float* W_ih1 = (const float*)d_in[1];
  const float* W_hh1 = (const float*)d_in[2];
  const float* b_ih1 = (const float*)d_in[3];
  const float* b_hh1 = (const float*)d_in[4];
  const float* W_ih2 = (const float*)d_in[5];
  const float* W_hh2 = (const float*)d_in[6];
  const float* b_ih2 = (const float*)d_in[7];
  const float* b_hh2 = (const float*)d_in[8];
  float* out = (float*)d_out;
  (void)in_sizes; (void)n_in; (void)out_size; (void)d_ws; (void)ws_size;

  // 4 workgroups: one 16-row batch tile each; 256 threads = 8 wave32.
  lstm2_scan_kernel<<<dim3(4), dim3(256), 0, stream>>>(
      x, W_ih1, W_hh1, b_ih1, b_hh1, W_ih2, W_hh2, b_ih2, b_hh2, out);
}
